// BaselineGVPModel_51651276702443
// MI455X (gfx1250) — compile-verified, matrix operand order checked by reading, not run on hardware
//
// GVP-GNN forward for MI455X (gfx1250, wave32, WMMA).
//
// Design (roofline): the per-edge message block dominates bytes, not FLOPs
// (~3.6 GB of gathers/scatters vs ~190 GFLOP). We therefore:
//  * fuse the 3 message GVPs per 16-edge tile (one wave each), LDS-resident
//  * gather node scalars as bf16 (halves gather traffic, feeds WMMA directly)
//  * run all wide ws_w GEMMs on v_wmma_f32_16x16x32_bf16 (fp32 accum)
//  * keep small vector-channel matmuls / norms / gates / LayerNorm in f32 VALU
//  * mean-aggregate with global_atomic_add_f32, pre-scaled by 1/deg
//
// Param flattening assumption: JAX pytree order (dict keys sorted):
//   6:We.wh 7:We.ws_b 8:We.ws_w 9:We.wv 10:WeLN.beta 11:WeLN.gamma
//   12:Wout.wh 13:Wout.ws_b 14:Wout.ws_w 15:WoutLN.beta 16:WoutLN.gamma
//   17:Wv.wh 18:Wv.ws_b 19:Wv.ws_w 20:Wv.wv 21:WvLN.beta 22:WvLN.gamma
//   23..28: cls w0,b0,w1,b1,w2,b2
//   layer l base = 29 + 24*l:
//     +0 ff0.wh +1 ff0.ws_b +2 ff0.ws_w +3 ff0.wv
//     +4 ff1.wh +5 ff1.ws_b +6 ff1.ws_w +7 ff1.wv
//     +8 ln0.beta +9 ln0.gamma +10 ln1.beta +11 ln1.gamma
//     +12 m0.wh +13 m0.ws_b +14 m0.ws_w +15 m0.wv
//     +16 m1.wh +17 m1.ws_b +18 m1.ws_w +19 m1.wv
//     +20 m2.wh +21 m2.ws_b +22 m2.ws_w +23 m2.wv

#include <hip/hip_runtime.h>
#include <math.h>

typedef __attribute__((ext_vector_type(16))) __bf16 v16bf;
typedef __attribute__((ext_vector_type(8)))  __bf16 v8bf;
typedef __attribute__((ext_vector_type(8)))  float  v8f;

#define DEV static __device__ __forceinline__

DEV __bf16 f2bf(float f) {
  union { float f; unsigned int u; } x; x.f = f;
  unsigned int u = x.u;
  u += 0x7FFFu + ((u >> 16) & 1u);          // round-to-nearest-even
  unsigned short hi = (unsigned short)(u >> 16);
  __bf16 b;
  __builtin_memcpy(&b, &hi, sizeof(b));
  return b;
}

// One 16x16 D tile: rows = 16 staged feature rows (LDS, bf16, stride kpad),
// cols = ntile-th group of 16 output channels from WT[Npad][wt_kpad] (global bf16).
// A layout (ISA 7.12.2): lane row = lane&15; component j -> k = 8*half + j + 8*(j>>3).
// B assumed symmetric (lane column = lane&15, same K mapping).
DEV v8f wmma_rowtile(const __bf16* feat, int kpad, int nchunks,
                     const __bf16* WT, int wt_kpad, int ntile) {
  const int lane = threadIdx.x & 31;
  const int l15  = lane & 15;
  const int half = lane >> 4;
  v8f acc = {};
  const __bf16* arow = feat + l15 * kpad;
  const __bf16* brow = WT + (size_t)(ntile * 16 + l15) * wt_kpad;
#pragma unroll
  for (int kc = 0; kc < nchunks; ++kc) {
    const int off = kc * 32 + 8 * half;
    v8bf a0 = *(const v8bf*)(arow + off);
    v8bf a1 = *(const v8bf*)(arow + off + 16);
    v8bf b0 = *(const v8bf*)(brow + off);
    v8bf b1 = *(const v8bf*)(brow + off + 16);
    v16bf a, b;
#pragma unroll
    for (int j = 0; j < 8; ++j) { a[j] = a0[j]; a[j + 8] = a1[j];
                                  b[j] = b0[j]; b[j + 8] = b1[j]; }
    acc = __builtin_amdgcn_wmma_f32_16x16x32_bf16(false, a, false, b,
                                                  (short)0, acc, false, false);
  }
  return acc;
}

// ---------------------------------------------------------------- utilities
__global__ void zero_kernel(float* p, size_t n) {
  size_t i = (size_t)blockIdx.x * blockDim.x + threadIdx.x;
  size_t stride = (size_t)gridDim.x * blockDim.x;
  for (; i < n; i += stride) p[i] = 0.f;
}

// fp32 W[K][N] -> bf16 WT[NPAD][KPAD] (transposed, zero padded)
__global__ void convert_wt_kernel(const float* __restrict__ w, __bf16* __restrict__ o,
                                  int K, int N, int KPAD, int NPAD) {
  int idx = blockIdx.x * blockDim.x + threadIdx.x;
  if (idx >= KPAD * NPAD) return;
  int nrow = idx / KPAD, k = idx - nrow * KPAD;
  float val = (k < K && nrow < N) ? w[(size_t)k * N + nrow] : 0.f;
  o[idx] = f2bf(val);
}

__global__ void degree_kernel(const int* __restrict__ dst, float* deg, int nE) {
  int e = blockIdx.x * blockDim.x + threadIdx.x;
  if (e < nE) atomicAdd(&deg[dst[e]], 1.f);
}
__global__ void invdeg_kernel(const float* __restrict__ deg, float* inv, int n) {
  int i = blockIdx.x * blockDim.x + threadIdx.x;
  if (i < n) inv[i] = 1.f / fmaxf(deg[i], 1.f);
}

// ---------------------------------------------------------------- embeddings
struct EmbNP {
  const float* hs; const float* hv;
  const float* g; const float* b;                 // ln gamma/beta (6)
  const float* wh; const float* wsw; const float* wsb; const float* wv;
  float* s; __bf16* s_bf; float* v; int n;
};
__global__ void embed_nodes_kernel(EmbNP p) {
  int i = blockIdx.x * blockDim.x + threadIdx.x;
  if (i >= p.n) return;
  const float* hs = p.hs + (size_t)i * 6;
  float feat[22];
  float mu = 0.f;
  for (int k = 0; k < 6; ++k) mu += hs[k];
  mu *= (1.f / 6.f);
  float var = 0.f;
  for (int k = 0; k < 6; ++k) { float d = hs[k] - mu; var += d * d; }
  var *= (1.f / 6.f);
  float inv = rsqrtf(var + 1e-5f);
  for (int k = 0; k < 6; ++k) feat[k] = (hs[k] - mu) * inv * p.g[k] + p.b[k];
  const float* hv = p.hv + (size_t)i * 9;
  float vs = 0.f;
  for (int vi = 0; vi < 3; ++vi) {
    const float* vp = hv + vi * 3;
    vs += fmaxf(vp[0]*vp[0] + vp[1]*vp[1] + vp[2]*vp[2], 1e-8f);
  }
  float r = rsqrtf(vs / 3.f);
  float vln[9];
  for (int j = 0; j < 9; ++j) vln[j] = hv[j] * r;
  float vh[48];
  for (int h = 0; h < 16; ++h) {
    float x = 0, y = 0, z = 0;
    for (int vi = 0; vi < 3; ++vi) {
      float w = p.wh[vi * 16 + h];
      x += vln[vi*3+0]*w; y += vln[vi*3+1]*w; z += vln[vi*3+2]*w;
    }
    vh[h*3]=x; vh[h*3+1]=y; vh[h*3+2]=z;
    feat[6 + h] = sqrtf(fmaxf(x*x + y*y + z*z, 1e-8f));
  }
  float* srow = p.s + (size_t)i * 100;
  for (int o = 0; o < 100; ++o) {
    float a = p.wsb[o];
    for (int k = 0; k < 22; ++k) a += feat[k] * p.wsw[k * 100 + o];
    srow[o] = a;
    p.s_bf[(size_t)i * 112 + o] = f2bf(a);
  }
  float* vrow = p.v + (size_t)i * 48;
  for (int vo = 0; vo < 16; ++vo) {
    float x = 0, y = 0, z = 0;
    for (int h = 0; h < 16; ++h) {
      float w = p.wv[h * 16 + vo];
      x += vh[h*3]*w; y += vh[h*3+1]*w; z += vh[h*3+2]*w;
    }
    vrow[vo*3]=x; vrow[vo*3+1]=y; vrow[vo*3+2]=z;
  }
}

struct EmbEP {
  const float* hs; const float* hv;
  const float* g; const float* b;
  const float* wh; const float* wsw; const float* wsb; const float* wv;
  float* es; float* ev; int n;
};
__global__ void embed_edges_kernel(EmbEP p) {
  int e = blockIdx.x * blockDim.x + threadIdx.x;
  if (e >= p.n) return;
  const float* hs = p.hs + (size_t)e * 32;
  float feat[33];
  float mu = 0.f;
  for (int k = 0; k < 32; ++k) mu += hs[k];
  mu *= (1.f / 32.f);
  float var = 0.f;
  for (int k = 0; k < 32; ++k) { float d = hs[k] - mu; var += d * d; }
  var *= (1.f / 32.f);
  float inv = rsqrtf(var + 1e-5f);
  for (int k = 0; k < 32; ++k) feat[k] = (hs[k] - mu) * inv * p.g[k] + p.b[k];
  const float* hv = p.hv + (size_t)e * 3;
  float ss = hv[0]*hv[0] + hv[1]*hv[1] + hv[2]*hv[2];
  float r = rsqrtf(fmaxf(ss, 1e-8f));
  float whs = p.wh[0], wvs = p.wv[0];
  float vh0 = hv[0]*r*whs, vh1 = hv[1]*r*whs, vh2 = hv[2]*r*whs;
  feat[32] = sqrtf(fmaxf(vh0*vh0 + vh1*vh1 + vh2*vh2, 1e-8f));
  float* eo = p.es + (size_t)e * 32;
  for (int o = 0; o < 32; ++o) {
    float a = p.wsb[o];
    for (int k = 0; k < 33; ++k) a += feat[k] * p.wsw[k * 32 + o];
    eo[o] = a;
  }
  float* vo = p.ev + (size_t)e * 3;
  vo[0] = vh0 * wvs; vo[1] = vh1 * wvs; vo[2] = vh2 * wvs;
}

// ------------------------------------------------------ fused message kernel
// one wave (32 threads) = tile of 16 edges; m0 -> m1 -> m2 fused in LDS.
struct MsgP {
  const int* src; const int* dst;
  const __bf16* s_bf; const float* v; const float* es; const float* ev;
  const float* inv_deg;
  float* acc_s; float* acc_v;
  const float* wh0; const float* wv0; const float* b0; const __bf16* wt0;
  const float* wh1; const float* wv1; const float* b1; const __bf16* wt1;
  const float* wh2; const float* wv2; const float* b2; const __bf16* wt2;
  int nE;
};

__global__ __launch_bounds__(32) void message_kernel(MsgP p) {
  __shared__ __align__(16) __bf16 feat0[16 * 288];   // m0 input (K=265 pad 288) / m2 input
  __shared__ __align__(16) __bf16 feat1[16 * 128];   // m1 input (K=116 pad 128)
  __shared__ float smv[16 * 33 * 3];
  __shared__ float svh[16 * 33 * 3];
  __shared__ float svc[16 * 16 * 3];
  __shared__ int   sdst[16];
  __shared__ float swdeg[16];
  __shared__ float w_wh0[33 * 33];
  __shared__ float w_wv0[33 * 16];
  __shared__ float w_wh1[256];
  __shared__ float w_wv1[256];
  __shared__ float w_wh2[256];
  __shared__ float w_wv2[256];

  const int lane = threadIdx.x;
  const int l15 = lane & 15, half = lane >> 4;
  const int ebase = blockIdx.x * 16;

  for (int i = lane; i < 33 * 33; i += 32) w_wh0[i] = p.wh0[i];
  for (int i = lane; i < 33 * 16; i += 32) w_wv0[i] = p.wv0[i];
  for (int i = lane; i < 256; i += 32) {
    w_wh1[i] = p.wh1[i]; w_wv1[i] = p.wv1[i];
    w_wh2[i] = p.wh2[i]; w_wv2[i] = p.wv2[i];
  }
  if (lane < 16) {
    int e = ebase + lane;
    bool valid = e < p.nE;
    int ec = valid ? e : 0;
    int d = p.dst[ec];
    sdst[lane] = d;
    swdeg[lane] = valid ? p.inv_deg[d] : 0.f;
  }
  // gather scalars -> feat0 (bf16): [s_src(100) | es(32) | s_dst(100) | vn0(33) | pad]
  for (int idx = lane; idx < 16 * 288; idx += 32) {
    int m = idx / 288, k = idx - m * 288;
    int e = ebase + m; if (e >= p.nE) e = 0;
    __bf16 val = f2bf(0.f);
    if (k < 100)      val = p.s_bf[(size_t)p.src[e] * 112 + k];
    else if (k < 132) val = f2bf(p.es[(size_t)e * 32 + (k - 100)]);
    else if (k < 232) val = p.s_bf[(size_t)p.dst[e] * 112 + (k - 132)];
    feat0[idx] = val;
  }
  // gather vectors -> smv: [v_src(16) | ev(1) | v_dst(16)] x xyz
  for (int idx = lane; idx < 16 * 33 * 3; idx += 32) {
    int m = idx / 99, r = idx - m * 99;
    int vi = r / 3, c = r - vi * 3;
    int e = ebase + m; if (e >= p.nE) e = 0;
    float val;
    if (vi < 16)       val = p.v[((size_t)p.src[e] * 16 + vi) * 3 + c];
    else if (vi == 16) val = p.ev[(size_t)e * 3 + c];
    else               val = p.v[((size_t)p.dst[e] * 16 + (vi - 17)) * 3 + c];
    smv[idx] = val;
  }
  __syncthreads();

  // m0: vh = mv @ wh0 (33->33), vn0 -> feat0[:,232+h]
  for (int t = lane; t < 16 * 33; t += 32) {
    int m = t / 33, h = t - m * 33;
    float x = 0, y = 0, z = 0;
    for (int vi = 0; vi < 33; ++vi) {
      float w = w_wh0[vi * 33 + h];
      const float* mp = &smv[(m * 33 + vi) * 3];
      x += mp[0]*w; y += mp[1]*w; z += mp[2]*w;
    }
    float* o = &svh[(m * 33 + h) * 3];
    o[0]=x; o[1]=y; o[2]=z;
    feat0[m * 288 + 232 + h] = f2bf(sqrtf(fmaxf(x*x + y*y + z*z, 1e-8f)));
  }
  for (int idx = lane; idx < 16 * 12; idx += 32) {    // pad feat1 cols 116..127
    int m = idx / 12, k = idx - m * 12;
    feat1[m * 128 + 116 + k] = f2bf(0.f);
  }
  __syncthreads();

  // m0 scalar GEMM (WMMA): [16x265]@[265x100] + b, relu -> feat1[:,0..99]
  for (int nt = 0; nt < 7; ++nt) {
    v8f acc = wmma_rowtile(feat0, 288, 9, p.wt0, 288, nt);
    int n = nt * 16 + l15;
    if (n < 100) {
      float b = p.b0[n];
#pragma unroll
      for (int r = 0; r < 8; ++r)
        feat1[(8 * half + r) * 128 + n] = f2bf(fmaxf(acc[r] + b, 0.f));
    }
  }
  __syncthreads();

  // m0 vector out + sigmoid gate -> svc
  for (int t = lane; t < 256; t += 32) {
    int m = t >> 4, vo = t & 15;
    float x = 0, y = 0, z = 0;
    for (int h = 0; h < 33; ++h) {
      float w = w_wv0[h * 16 + vo];
      const float* vp = &svh[(m * 33 + h) * 3];
      x += vp[0]*w; y += vp[1]*w; z += vp[2]*w;
    }
    float g = 1.f / (1.f + __expf(-sqrtf(fmaxf(x*x + y*y + z*z, 1e-8f))));
    float* o = &svc[(m * 16 + vo) * 3];
    o[0]=x*g; o[1]=y*g; o[2]=z*g;
  }
  __syncthreads();

  // m1: vh1 = v1 @ wh1 (16->16) -> svh (stride 16), vn1 -> feat1[:,100+h]
  for (int t = lane; t < 256; t += 32) {
    int m = t >> 4, h = t & 15;
    float x = 0, y = 0, z = 0;
    for (int vi = 0; vi < 16; ++vi) {
      float w = w_wh1[vi * 16 + h];
      const float* vp = &svc[(m * 16 + vi) * 3];
      x += vp[0]*w; y += vp[1]*w; z += vp[2]*w;
    }
    float* o = &svh[(m * 16 + h) * 3];
    o[0]=x; o[1]=y; o[2]=z;
    feat1[m * 128 + 100 + h] = f2bf(sqrtf(fmaxf(x*x + y*y + z*z, 1e-8f)));
  }
  for (int idx = lane; idx < 16 * 12; idx += 32) {    // pad feat0 cols 116..127
    int m = idx / 12, k = idx - m * 12;
    feat0[m * 288 + 116 + k] = f2bf(0.f);
  }
  __syncthreads();

  // m1 scalar GEMM -> relu -> feat0[:,0..99]
  for (int nt = 0; nt < 7; ++nt) {
    v8f acc = wmma_rowtile(feat1, 128, 4, p.wt1, 128, nt);
    int n = nt * 16 + l15;
    if (n < 100) {
      float b = p.b1[n];
#pragma unroll
      for (int r = 0; r < 8; ++r)
        feat0[(8 * half + r) * 288 + n] = f2bf(fmaxf(acc[r] + b, 0.f));
    }
  }
  __syncthreads();

  // m1 vector out + gate -> svc
  for (int t = lane; t < 256; t += 32) {
    int m = t >> 4, vo = t & 15;
    float x = 0, y = 0, z = 0;
    for (int h = 0; h < 16; ++h) {
      float w = w_wv1[h * 16 + vo];
      const float* vp = &svh[(m * 16 + h) * 3];
      x += vp[0]*w; y += vp[1]*w; z += vp[2]*w;
    }
    float g = 1.f / (1.f + __expf(-sqrtf(fmaxf(x*x + y*y + z*z, 1e-8f))));
    float* o = &svc[(m * 16 + vo) * 3];
    o[0]=x*g; o[1]=y*g; o[2]=z*g;
  }
  __syncthreads();

  // m2: vh2 -> svh, vn2 -> feat0[:,100+h]
  for (int t = lane; t < 256; t += 32) {
    int m = t >> 4, h = t & 15;
    float x = 0, y = 0, z = 0;
    for (int vi = 0; vi < 16; ++vi) {
      float w = w_wh2[vi * 16 + h];
      const float* vp = &svc[(m * 16 + vi) * 3];
      x += vp[0]*w; y += vp[1]*w; z += vp[2]*w;
    }
    float* o = &svh[(m * 16 + h) * 3];
    o[0]=x; o[1]=y; o[2]=z;
    feat0[m * 288 + 100 + h] = f2bf(sqrtf(fmaxf(x*x + y*y + z*z, 1e-8f)));
  }
  __syncthreads();

  // m2 scalar GEMM (no act) -> mean-aggregated atomic add into acc_s
  for (int nt = 0; nt < 7; ++nt) {
    v8f acc = wmma_rowtile(feat0, 288, 4, p.wt2, 128, nt);
    int n = nt * 16 + l15;
    if (n < 100) {
      float b = p.b2[n];
#pragma unroll
      for (int r = 0; r < 8; ++r) {
        int m = 8 * half + r;
        float w = swdeg[m];
        if (w != 0.f)
          atomicAdd(&p.acc_s[(size_t)sdst[m] * 100 + n], (acc[r] + b) * w);
      }
    }
  }
  // m2 vector out (no gate) -> atomic add into acc_v
  for (int t = lane; t < 256; t += 32) {
    int m = t >> 4, vo = t & 15;
    float x = 0, y = 0, z = 0;
    for (int h = 0; h < 16; ++h) {
      float w = w_wv2[h * 16 + vo];
      const float* vp = &svh[(m * 16 + h) * 3];
      x += vp[0]*w; y += vp[1]*w; z += vp[2]*w;
    }
    float w = swdeg[m];
    if (w != 0.f) {
      float* bp = &p.acc_v[((size_t)sdst[m] * 16 + vo) * 3];
      atomicAdd(bp + 0, x * w);
      atomicAdd(bp + 1, y * w);
      atomicAdd(bp + 2, z * w);
    }
  }
}

// ------------------------------------------- residual + LayerNorm (per node)
struct NodeUpP {
  float* s; float* v; __bf16* s_bf;
  const float* acc_s; const float* acc_v;
  const float* gamma; const float* beta;
  int n;
};
__global__ void node_update_kernel(NodeUpP p) {
  int i = blockIdx.x * blockDim.x + threadIdx.x;
  if (i >= p.n) return;
  float tmp[100];
  const float* sr = p.s + (size_t)i * 100;
  const float* ar = p.acc_s + (size_t)i * 100;
  float mu = 0.f;
  for (int k = 0; k < 100; ++k) { tmp[k] = sr[k] + ar[k]; mu += tmp[k]; }
  mu *= 0.01f;
  float var = 0.f;
  for (int k = 0; k < 100; ++k) { float d = tmp[k] - mu; var += d * d; }
  var *= 0.01f;
  float inv = rsqrtf(var + 1e-5f);
  float* so = p.s + (size_t)i * 100;
  for (int k = 0; k < 100; ++k) {
    float val = (tmp[k] - mu) * inv * p.gamma[k] + p.beta[k];
    so[k] = val;
    p.s_bf[(size_t)i * 112 + k] = f2bf(val);
  }
  float vv[48];
  const float* vr = p.v + (size_t)i * 48;
  const float* avr = p.acc_v + (size_t)i * 48;
  float vs = 0.f;
  for (int vo = 0; vo < 16; ++vo) {
    float a = vr[vo*3+0] + avr[vo*3+0];
    float b = vr[vo*3+1] + avr[vo*3+1];
    float c = vr[vo*3+2] + avr[vo*3+2];
    vv[vo*3]=a; vv[vo*3+1]=b; vv[vo*3+2]=c;
    vs += fmaxf(a*a + b*b + c*c, 1e-8f);
  }
  float r = rsqrtf(vs / 16.f);
  float* vo_ = p.v + (size_t)i * 48;
  for (int j = 0; j < 48; ++j) vo_[j] = vv[j] * r;
}

// --------------------------------------------------- fused feed-forward (FF)
struct FFP {
  const float* s; const float* v; const __bf16* s_bf;
  float* acc_s; float* acc_v;
  const float* wh0; const float* wv0; const float* b0; const __bf16* wt0;
  const float* wh1; const float* wv1; const float* b1; const __bf16* wt1;
  int n;
};
__global__ __launch_bounds__(32) void ff_kernel(FFP p) {
  __shared__ __align__(16) __bf16 featA[16 * 160];   // ff0 input (K=132 pad 160)
  __shared__ __align__(16) __bf16 featB[16 * 448];   // ff1 input (K=432 pad 448)
  __shared__ float vloc[16 * 16 * 3];
  __shared__ float vhA[16 * 32 * 3];
  __shared__ float vtmp[16 * 32 * 3];
  __shared__ float w_wh0[16 * 32];
  __shared__ float w_wv0[32 * 32];
  __shared__ float w_wh1[32 * 32];
  __shared__ float w_wv1[32 * 16];

  const int lane = threadIdx.x;
  const int l15 = lane & 15, half = lane >> 4;
  const int nbase = blockIdx.x * 16;

  for (int i = lane; i < 512;  i += 32) w_wh0[i] = p.wh0[i];
  for (int i = lane; i < 1024; i += 32) w_wv0[i] = p.wv0[i];
  for (int i = lane; i < 1024; i += 32) w_wh1[i] = p.wh1[i];
  for (int i = lane; i < 512;  i += 32) w_wv1[i] = p.wv1[i];

  for (int idx = lane; idx < 16 * 160; idx += 32) {
    int m = idx / 160, k = idx - m * 160;
    int node = nbase + m; if (node >= p.n) node = 0;
    featA[idx] = (k < 100) ? p.s_bf[(size_t)node * 112 + k] : f2bf(0.f);
  }
  for (int idx = lane; idx < 16 * 48; idx += 32) {
    int m = idx / 48;
    int node = nbase + m; if (node >= p.n) node = 0;
    vloc[idx] = p.v[(size_t)node * 48 + (idx - m * 48)];
  }
  __syncthreads();

  // ff0: vhA = v @ wh (16->32), vn -> featA[:,100..131]
  for (int t = lane; t < 16 * 32; t += 32) {
    int m = t >> 5, h = t & 31;
    float x = 0, y = 0, z = 0;
    for (int vi = 0; vi < 16; ++vi) {
      float w = w_wh0[vi * 32 + h];
      const float* vp = &vloc[(m * 16 + vi) * 3];
      x += vp[0]*w; y += vp[1]*w; z += vp[2]*w;
    }
    float* o = &vhA[(m * 32 + h) * 3];
    o[0]=x; o[1]=y; o[2]=z;
    featA[m * 160 + 100 + h] = f2bf(sqrtf(fmaxf(x*x + y*y + z*z, 1e-8f)));
  }
  for (int idx = lane; idx < 16 * 16; idx += 32) {    // pad featB cols 432..447
    int m = idx >> 4, k = idx & 15;
    featB[m * 448 + 432 + k] = f2bf(0.f);
  }
  __syncthreads();

  // ff0 scalar GEMM: [16x132]@[132x400] + b, relu -> featB[:,0..399]
  for (int nt = 0; nt < 25; ++nt) {
    v8f acc = wmma_rowtile(featA, 160, 5, p.wt0, 160, nt);
    int n = nt * 16 + l15;                              // < 400 always
    float b = p.b0[n];
#pragma unroll
    for (int r = 0; r < 8; ++r)
      featB[(8 * half + r) * 448 + n] = f2bf(fmaxf(acc[r] + b, 0.f));
  }
  __syncthreads();

  // ff0 vector out + gate -> vtmp
  for (int t = lane; t < 16 * 32; t += 32) {
    int m = t >> 5, vo = t & 31;
    float x = 0, y = 0, z = 0;
    for (int h = 0; h < 32; ++h) {
      float w = w_wv0[h * 32 + vo];
      const float* vp = &vhA[(m * 32 + h) * 3];
      x += vp[0]*w; y += vp[1]*w; z += vp[2]*w;
    }
    float g = 1.f / (1.f + __expf(-sqrtf(fmaxf(x*x + y*y + z*z, 1e-8f))));
    float* o = &vtmp[(m * 32 + vo) * 3];
    o[0]=x*g; o[1]=y*g; o[2]=z*g;
  }
  __syncthreads();

  // ff1: vhA = vtmp @ wh1 (32->32), vn -> featB[:,400..431]
  for (int t = lane; t < 16 * 32; t += 32) {
    int m = t >> 5, h = t & 31;
    float x = 0, y = 0, z = 0;
    for (int vi = 0; vi < 32; ++vi) {
      float w = w_wh1[vi * 32 + h];
      const float* vp = &vtmp[(m * 32 + vi) * 3];
      x += vp[0]*w; y += vp[1]*w; z += vp[2]*w;
    }
    float* o = &vhA[(m * 32 + h) * 3];
    o[0]=x; o[1]=y; o[2]=z;
    featB[m * 448 + 400 + h] = f2bf(sqrtf(fmaxf(x*x + y*y + z*z, 1e-8f)));
  }
  __syncthreads();

  // ff1 scalar GEMM (no act): fs -> acc_s
  for (int nt = 0; nt < 7; ++nt) {
    v8f acc = wmma_rowtile(featB, 448, 14, p.wt1, 448, nt);
    int n = nt * 16 + l15;
    if (n < 100) {
      float b = p.b1[n];
#pragma unroll
      for (int r = 0; r < 8; ++r) {
        int node = nbase + 8 * half + r;
        if (node < p.n) p.acc_s[(size_t)node * 100 + n] = acc[r] + b;
      }
    }
  }
  // ff1 vector out (no act): fv -> acc_v
  for (int t = lane; t < 256; t += 32) {
    int m = t >> 4, vo = t & 15;
    float x = 0, y = 0, z = 0;
    for (int h = 0; h < 32; ++h) {
      float w = w_wv1[h * 16 + vo];
      const float* vp = &vhA[(m * 32 + h) * 3];
      x += vp[0]*w; y += vp[1]*w; z += vp[2]*w;
    }
    int node = nbase + m;
    if (node < p.n) {
      float* o = &p.acc_v[((size_t)node * 16 + vo) * 3];
      o[0]=x; o[1]=y; o[2]=z;
    }
  }
}

// ------------------------------------------ output head + mean pool per graph
struct OutP {
  const float* s; const float* v;
  const float* ln_g; const float* ln_b;
  const float* wh; const float* bias; const __bf16* wt;
  const int* batch;
  float* gsum; float* gcnt;
  int n;
};
__global__ __launch_bounds__(32) void out_pool_kernel(OutP p) {
  __shared__ __align__(16) __bf16 featC[16 * 128];
  __shared__ float vloc[16 * 16 * 3];
  __shared__ float w_wh[256];
  __shared__ int gid[16];

  const int lane = threadIdx.x;
  const int l15 = lane & 15, half = lane >> 4;
  const int nbase = blockIdx.x * 16;

  for (int i = lane; i < 256; i += 32) w_wh[i] = p.wh[i];
  if (lane < 16) {
    int node = nbase + lane;
    bool valid = node < p.n;
    int nc = valid ? node : 0;
    int g = p.batch[nc];
    gid[lane] = valid ? g : -1;
    if (valid) atomicAdd(&p.gcnt[g], 1.f);
    const float* srow = p.s + (size_t)nc * 100;
    float mu = 0.f;
    for (int k = 0; k < 100; ++k) mu += srow[k];
    mu *= 0.01f;
    float var = 0.f;
    for (int k = 0; k < 100; ++k) { float d = srow[k] - mu; var += d * d; }
    var *= 0.01f;
    float inv = rsqrtf(var + 1e-5f);
    for (int k = 0; k < 100; ++k)
      featC[lane * 128 + k] = f2bf((srow[k] - mu) * inv * p.ln_g[k] + p.ln_b[k]);
    for (int k = 116; k < 128; ++k) featC[lane * 128 + k] = f2bf(0.f);
    const float* vrow = p.v + (size_t)nc * 48;
    float vs = 0.f;
    for (int vo = 0; vo < 16; ++vo) {
      const float* vp = vrow + vo * 3;
      vs += fmaxf(vp[0]*vp[0] + vp[1]*vp[1] + vp[2]*vp[2], 1e-8f);
    }
    float r = rsqrtf(vs / 16.f);
    for (int j = 0; j < 48; ++j) vloc[lane * 48 + j] = vrow[j] * r;
  }
  __syncthreads();
  for (int t = lane; t < 256; t += 32) {              // vn -> featC[:,100..115]
    int m = t >> 4, h = t & 15;
    float x = 0, y = 0, z = 0;
    for (int vi = 0; vi < 16; ++vi) {
      float w = w_wh[vi * 16 + h];
      const float* vp = &vloc[(m * 16 + vi) * 3];
      x += vp[0]*w; y += vp[1]*w; z += vp[2]*w;
    }
    featC[m * 128 + 100 + h] = f2bf(sqrtf(fmaxf(x*x + y*y + z*z, 1e-8f)));
  }
  __syncthreads();
  for (int nt = 0; nt < 7; ++nt) {
    v8f acc = wmma_rowtile(featC, 128, 4, p.wt, 128, nt);
    int n = nt * 16 + l15;
    if (n < 100) {
      float b = p.bias[n];
#pragma unroll
      for (int r = 0; r < 8; ++r) {
        int g = gid[8 * half + r];
        if (g >= 0) atomicAdd(&p.gsum[(size_t)g * 100 + n], fmaxf(acc[r] + b, 0.f));
      }
    }
  }
}

// ------------------------------------------------------------ classifier MLP
struct ClsP {
  const float* gsum; const float* gcnt;
  const float* w0; const float* b0;
  const float* w1; const float* b1;
  const float* w2; const float* b2;
  float* out; int nG;
};
__global__ __launch_bounds__(256) void classifier_kernel(ClsP p) {
  __shared__ float g[16 * 100];
  __shared__ float h1[16 * 200];
  __shared__ float h2[16 * 100];
  int tid = threadIdx.x;
  for (int i = tid; i < p.nG * 100; i += 256)
    g[i] = p.gsum[i] / fmaxf(p.gcnt[i / 100], 1.f);
  __syncthreads();
  for (int i = tid; i < p.nG * 200; i += 256) {
    int gi = i / 200, o = i - gi * 200;
    float a = p.b0[o];
    for (int k = 0; k < 100; ++k) a += g[gi * 100 + k] * p.w0[k * 200 + o];
    h1[i] = fmaxf(a, 0.f);
  }
  __syncthreads();
  for (int i = tid; i < p.nG * 100; i += 256) {
    int gi = i / 100, o = i - gi * 100;
    float a = p.b1[o];
    for (int k = 0; k < 200; ++k) a += h1[gi * 200 + k] * p.w1[k * 100 + o];
    h2[i] = fmaxf(a, 0.f);
  }
  __syncthreads();
  for (int i = tid; i < p.nG * 10; i += 256) {
    int gi = i / 10, o = i - gi * 10;
    float a = p.b2[o];
    for (int k = 0; k < 100; ++k) a += h2[gi * 100 + k] * p.w2[k * 10 + o];
    p.out[i] = a;
  }
}

// --------------------------------------------------------------------- host
extern "C" void kernel_launch(void* const* d_in, const int* in_sizes, int n_in,
                              void* d_out, int out_size, void* d_ws, size_t ws_size,
                              hipStream_t stream) {
  const int nN = in_sizes[0] / 6;     // 20000
  const int nE = in_sizes[2] / 2;     // 600000
  const int nG = out_size / 10;       // 16
  (void)n_in; (void)ws_size;

  const float* hVs = (const float*)d_in[0];
  const float* hVv = (const float*)d_in[1];
  const int*   eidx = (const int*)d_in[2];
  const float* hEs = (const float*)d_in[3];
  const float* hEv = (const float*)d_in[4];
  const int*   batch = (const int*)d_in[5];
  const int* esrc = eidx;
  const int* edst = eidx + nE;
  auto P = [&](int i) -> const float* { return (const float*)d_in[i]; };

  // workspace carve (256B aligned)
  char* ws = (char*)d_ws;
  size_t cur = 0;
  auto take = [&](size_t bytes) -> void* {
    void* ptr = ws + cur;
    cur += (bytes + 255) & ~(size_t)255;
    return ptr;
  };
  float*  s_f32  = (float*) take((size_t)nN * 100 * 4);
  float*  v_f32  = (float*) take((size_t)nN * 48 * 4);
  __bf16* s_bf   = (__bf16*)take((size_t)nN * 112 * 2);
  float*  es     = (float*) take((size_t)nE * 32 * 4);
  float*  ev     = (float*) take((size_t)nE * 3 * 4);
  float*  acc_s  = (float*) take((size_t)nN * 100 * 4);
  float*  acc_v  = (float*) take((size_t)nN * 48 * 4);
  float*  deg    = (float*) take((size_t)nN * 4);
  float*  invdeg = (float*) take((size_t)nN * 4);
  float*  gsum   = (float*) take((size_t)nG * 100 * 4);
  float*  gcnt   = (float*) take((size_t)nG * 4);
  __bf16* wt_m0[3]; __bf16* wt_m1[3]; __bf16* wt_m2[3];
  __bf16* wt_ff0[3]; __bf16* wt_ff1[3];
  for (int l = 0; l < 3; ++l) {
    wt_m0[l]  = (__bf16*)take((size_t)112 * 288 * 2);
    wt_m1[l]  = (__bf16*)take((size_t)112 * 128 * 2);
    wt_m2[l]  = (__bf16*)take((size_t)112 * 128 * 2);
    wt_ff0[l] = (__bf16*)take((size_t)400 * 160 * 2);
    wt_ff1[l] = (__bf16*)take((size_t)112 * 448 * 2);
  }
  __bf16* wt_out = (__bf16*)take((size_t)112 * 128 * 2);

  auto conv = [&](const float* w, __bf16* o, int K, int N, int KPAD, int NPAD) {
    int total = KPAD * NPAD;
    convert_wt_kernel<<<(total + 255) / 256, 256, 0, stream>>>(w, o, K, N, KPAD, NPAD);
  };
  for (int l = 0; l < 3; ++l) {
    int b = 29 + 24 * l;
    conv(P(b + 14), wt_m0[l], 265, 100, 288, 112);
    conv(P(b + 18), wt_m1[l], 116, 100, 128, 112);
    conv(P(b + 22), wt_m2[l], 116, 100, 128, 112);
    conv(P(b + 2),  wt_ff0[l], 132, 400, 160, 400);
    conv(P(b + 6),  wt_ff1[l], 432, 100, 448, 112);
  }
  conv(P(14), wt_out, 116, 100, 128, 112);

  auto zero = [&](float* pp, size_t n) {
    size_t blocks = (n + 255) / 256;
    if (blocks > 4096) blocks = 4096;
    zero_kernel<<<dim3((unsigned)blocks), 256, 0, stream>>>(pp, n);
  };
  zero((float*)s_bf, (size_t)nN * 56);   // nN*112 bf16 == nN*56 floats
  zero(deg, (size_t)nN);

  EmbNP en{hVs, hVv, P(22), P(21), P(17), P(19), P(18), P(20), s_f32, s_bf, v_f32, nN};
  embed_nodes_kernel<<<(nN + 127) / 128, 128, 0, stream>>>(en);
  EmbEP ee{hEs, hEv, P(11), P(10), P(6), P(8), P(7), P(9), es, ev, nE};
  embed_edges_kernel<<<(nE + 127) / 128, 128, 0, stream>>>(ee);
  degree_kernel<<<(nE + 255) / 256, 256, 0, stream>>>(edst, deg, nE);
  invdeg_kernel<<<(nN + 255) / 256, 256, 0, stream>>>(deg, invdeg, nN);

  for (int l = 0; l < 3; ++l) {
    int b = 29 + 24 * l;
    zero(acc_s, (size_t)nN * 100);
    zero(acc_v, (size_t)nN * 48);
    MsgP mp;
    mp.src = esrc; mp.dst = edst; mp.s_bf = s_bf; mp.v = v_f32;
    mp.es = es; mp.ev = ev; mp.inv_deg = invdeg;
    mp.acc_s = acc_s; mp.acc_v = acc_v;
    mp.wh0 = P(b + 12); mp.wv0 = P(b + 15); mp.b0 = P(b + 13); mp.wt0 = wt_m0[l];
    mp.wh1 = P(b + 16); mp.wv1 = P(b + 19); mp.b1 = P(b + 17); mp.wt1 = wt_m1[l];
    mp.wh2 = P(b + 20); mp.wv2 = P(b + 23); mp.b2 = P(b + 21); mp.wt2 = wt_m2[l];
    mp.nE = nE;
    message_kernel<<<(nE + 15) / 16, 32, 0, stream>>>(mp);

    NodeUpP nu0{s_f32, v_f32, s_bf, acc_s, acc_v, P(b + 9), P(b + 8), nN};
    node_update_kernel<<<(nN + 127) / 128, 128, 0, stream>>>(nu0);

    FFP fp;
    fp.s = s_f32; fp.v = v_f32; fp.s_bf = s_bf; fp.acc_s = acc_s; fp.acc_v = acc_v;
    fp.wh0 = P(b + 0); fp.wv0 = P(b + 3); fp.b0 = P(b + 1); fp.wt0 = wt_ff0[l];
    fp.wh1 = P(b + 4); fp.wv1 = P(b + 7); fp.b1 = P(b + 5); fp.wt1 = wt_ff1[l];
    fp.n = nN;
    ff_kernel<<<(nN + 15) / 16, 32, 0, stream>>>(fp);

    NodeUpP nu1{s_f32, v_f32, s_bf, acc_s, acc_v, P(b + 11), P(b + 10), nN};
    node_update_kernel<<<(nN + 127) / 128, 128, 0, stream>>>(nu1);
  }

  zero(gsum, (size_t)nG * 100);
  zero(gcnt, (size_t)nG);
  OutP op{s_f32, v_f32, P(16), P(15), P(12), P(13), wt_out, batch, gsum, gcnt, nN};
  out_pool_kernel<<<(nN + 15) / 16, 32, 0, stream>>>(op);
  ClsP cp{gsum, gcnt, P(23), P(24), P(25), P(26), P(27), P(28), (float*)d_out, nG};
  classifier_kernel<<<1, 256, 0, stream>>>(cp);
}
